// VectorQuantizerEMA_25288767439191
// MI455X (gfx1250) — matrix operand order repaired and checked by reading, not compiled, not run on hardware
//
#include <hip/hip_runtime.h>

typedef __attribute__((ext_vector_type(16))) _Float16 v16h;
typedef __attribute__((ext_vector_type(8)))  _Float16 v8h;
typedef __attribute__((ext_vector_type(4)))  _Float16 v4h;
typedef __attribute__((ext_vector_type(8)))  float    v8f;

#define N_TOK   32768
#define KCB     4096
#define DIM     128
#define ROWS_WG 128
#define NTILE   (KCB / 16)
#define DECAYF  0.99f
#define ONEMD   0.01f
#define EPSF    1e-5f

union hv { v16h v; v8h h[2]; };

// LDS byte offset of a __shared__ object: generic (flat) pointers to LDS are
// aperture|offset, so the low 32 bits are the LDS address.
__device__ __forceinline__ unsigned lds_addr(const void* p) {
  return (unsigned)(unsigned long long)p;
}

// Async global->LDS 16-byte copy (GVS mode), tracked by ASYNCcnt.
__device__ __forceinline__ void async_copy16(unsigned lds_off, unsigned g_off,
                                             unsigned long long g_base) {
  asm volatile("global_load_async_to_lds_b128 %0, %1, %2"
               :: "v"(lds_off), "v"(g_off), "s"(g_base) : "memory");
}
__device__ __forceinline__ void wait_async0() {
  asm volatile("s_wait_asynccnt 0x0" ::: "memory");
}

// ---------------------------------------------------------------------------
// Kernel 0: f32 -> f16 copy (X and codebook) so hot-loop staging is byte-pure.
// ---------------------------------------------------------------------------
__global__ __launch_bounds__(256)
void k_cvt(const float* __restrict__ src, _Float16* __restrict__ dst) {
  int i = blockIdx.x * 256 + threadIdx.x;
  float4 v = ((const float4*)src)[i];
  v4h o;
  o.x = (_Float16)v.x; o.y = (_Float16)v.y;
  o.z = (_Float16)v.z; o.w = (_Float16)v.w;
  ((v4h*)dst)[i] = o;
}

// ---------------------------------------------------------------------------
// Kernel 1: per-code squared norms (exact f32) + zero workspace accumulators.
// ---------------------------------------------------------------------------
__global__ __launch_bounds__(128)
void k_init(const float* __restrict__ CB, float* __restrict__ cnorm,
            float* __restrict__ soft_ws, float* __restrict__ bcount,
            float* __restrict__ bweight, float* __restrict__ total) {
  int k = blockIdx.x, d = threadIdx.x;
  float c = CB[k * DIM + d];
  bweight[k * DIM + d] = 0.0f;
  float s = c * c;
  #pragma unroll
  for (int m = 16; m >= 1; m >>= 1) s += __shfl_xor(s, m, 32);
  __shared__ float red[4];
  if ((d & 31) == 0) red[d >> 5] = s;
  __syncthreads();
  if (d == 0) {
    cnorm[k]   = red[0] + red[1] + red[2] + red[3];
    soft_ws[k] = 0.0f;
    bcount[k]  = 0.0f;
    if (k == 0) *total = 0.0f;
  }
}

// ---------------------------------------------------------------------------
// Kernel 2: fused distance GEMM (v_wmma_f32_16x16x32_f16) + softmax + argmin.
// 128 rows/WG (8 waves x 16-row slabs). Codebook swept in 16-col tiles that
// are DOUBLE-BUFFERED via global_load_async_to_lds_b128 (ASYNCcnt), so tile
// t+1 streams into LDS while WMMAs consume tile t. Softmax/argmin stats are
// lane-local in the hot loop; one butterfly merge per pass.
// ---------------------------------------------------------------------------
__global__ __launch_bounds__(256)
void k_main(const _Float16* __restrict__ Xh, const _Float16* __restrict__ Ch,
            const float* __restrict__ cnorm, int* __restrict__ codes,
            float* __restrict__ soft_g) {
  __shared__ __align__(16) _Float16 lds_x[ROWS_WG * DIM];  // 32 KB X tile
  __shared__ __align__(16) _Float16 lds_c[2][16 * DIM];    // 2 x 4 KB C tiles
  __shared__ float lds_soft[KCB];                          // 16 KB soft acc

  const int tid  = threadIdx.x;
  const int wave = tid >> 5;
  const int lane = tid & 31;
  const int half = lane >> 4;
  const int l16  = lane & 15;
  const int rowbase = blockIdx.x * ROWS_WG;
  const int wrow = wave * 16;
  const float inv_scale = 0.08838834764831845f; // 1/sqrt(128)

  const unsigned long long xg = (unsigned long long)(Xh + (size_t)rowbase * DIM);
  const unsigned long long cg = (unsigned long long)Ch;
  const unsigned lx = lds_addr(&lds_x[0]);
  const unsigned lc0 = lds_addr(&lds_c[0][0]);
  const unsigned lc1 = lds_addr(&lds_c[1][0]);

  // Issue one C-tile copy: 16 cols x 128 d x 2B = 4 KB = 1 b128 per thread.
  auto issue_c = [&](int t, int bufsel) {
    unsigned loff = (bufsel ? lc1 : lc0) + (unsigned)(tid * 16);
    unsigned goff = (unsigned)(t * (16 * DIM * 2) + tid * 16);
    async_copy16(loff, goff, cg);
  };

  // Prologue: async-stage X tile (32 KB = 8 b128/thread) + first C tile,
  // zero the soft accumulator meanwhile.
  #pragma unroll
  for (int k = 0; k < 8; ++k)
    async_copy16(lx + (unsigned)(tid * 16 + k * 4096),
                 (unsigned)(tid * 16 + k * 4096), xg);
  issue_c(0, 0);
  for (int i = tid; i < KCB; i += 256) lds_soft[i] = 0.0f;
  wait_async0();
  __syncthreads();

  // A fragments (wave's 16 rows x 128 K) hoisted to registers.
  // 16-bit A layout: lanes 0-15 K {0..7,16..23}; lanes 16-31 K {8..15,24..31}.
  v16h afrag[4];
  {
    const int arow = wrow + l16;
    #pragma unroll
    for (int c = 0; c < 4; ++c) {
      int d0 = c * 32 + (half ? 8 : 0);
      hv u;
      u.h[0] = *(const v8h*)&lds_x[arow * DIM + d0];
      u.h[1] = *(const v8h*)&lds_x[arow * DIM + d0 + 16];
      afrag[c] = u.v;
    }
  }

  // Lane-local stats per row slot v (row = wrow + v + 8*half).
  // Logits are O(1e-2) here so exp() needs no baseline shift.
  float Ml[8], Zl[8];
  int bi[8];
  #pragma unroll
  for (int v = 0; v < 8; ++v) { Ml[v] = -3.0e38f; Zl[v] = 0.0f; bi[v] = 0; }

  // ---------------- PASS 1: max/argmax + normalizer ----------------
  for (int t = 0; t < NTILE; ++t) {
    wait_async0();       // this wave's copies for tile t are complete
    __syncthreads();     // all waves' copies done; all done computing t-1
    if (t + 1 < NTILE) issue_c(t + 1, (t + 1) & 1);

    const _Float16* cbuf = &lds_c[t & 1][0];
    const int kcol = t * 16 + l16;
    const float cn = cnorm[kcol];
    // 16-bit B layout: lanes 0-15 K {0..7,16..23}; lanes 16-31 K {8..15,24..31}.
    hv ub[4];
    const int cb0 = l16 * DIM + half * 8;
    #pragma unroll
    for (int c = 0; c < 4; ++c) {
      ub[c].h[0] = *(const v8h*)&cbuf[cb0 + c * 32];
      ub[c].h[1] = *(const v8h*)&cbuf[cb0 + c * 32 + 16];
    }
    v8f acc = {0.f, 0.f, 0.f, 0.f, 0.f, 0.f, 0.f, 0.f};
    #pragma unroll
    for (int c = 0; c < 4; ++c)
      acc = __builtin_amdgcn_wmma_f32_16x16x32_f16(
          false, afrag[c], false, ub[c].v, (short)0, acc, false, false);

    #pragma unroll
    for (int v = 0; v < 8; ++v) {
      float l = __fmaf_rn(2.0f, acc[v], -cn) * inv_scale;
      Zl[v] += __expf(l);
      bi[v] = (l > Ml[v]) ? kcol : bi[v];   // strict > keeps earliest k
      Ml[v] = fmaxf(Ml[v], l);
    }
  }

  // Merge 16 lane-partials per row: argmax (tie -> min k) and Z sum.
  #pragma unroll
  for (int v = 0; v < 8; ++v) {
    float M = Ml[v], Zs = Zl[v];
    int ti = bi[v];
    #pragma unroll
    for (int m = 8; m >= 1; m >>= 1) {
      float oM = __shfl_xor(M, m, 16);
      float oZ = __shfl_xor(Zs, m, 16);
      int   oi = __shfl_xor(ti, m, 16);
      Zs += oZ;
      int take = (int)(oM > M) | ((int)(oM == M) & (int)(oi < ti));
      ti = take ? oi : ti;
      M  = fmaxf(M, oM);
    }
    Zl[v] = 1.0f / Zs;  // reuse as invZ for pass 2
    bi[v] = ti;
  }
  if (l16 == 0) {
    #pragma unroll
    for (int v = 0; v < 8; ++v)
      codes[rowbase + wrow + half * 8 + v] = bi[v];
  }

  // ---------------- PASS 2: soft_usage accumulation ----------------
  issue_c(0, 0);  // safe: all waves past reading buf0 (last read at t=NTILE-2)
  for (int t = 0; t < NTILE; ++t) {
    wait_async0();
    __syncthreads();
    if (t + 1 < NTILE) issue_c(t + 1, (t + 1) & 1);

    const _Float16* cbuf = &lds_c[t & 1][0];
    const int kcol = t * 16 + l16;
    const float cn = cnorm[kcol];
    hv ub[4];
    const int cb0 = l16 * DIM + half * 8;
    #pragma unroll
    for (int c = 0; c < 4; ++c) {
      ub[c].h[0] = *(const v8h*)&cbuf[cb0 + c * 32];
      ub[c].h[1] = *(const v8h*)&cbuf[cb0 + c * 32 + 16];
    }
    v8f acc = {0.f, 0.f, 0.f, 0.f, 0.f, 0.f, 0.f, 0.f};
    #pragma unroll
    for (int c = 0; c < 4; ++c)
      acc = __builtin_amdgcn_wmma_f32_16x16x32_f16(
          false, afrag[c], false, ub[c].v, (short)0, acc, false, false);

    float psum = 0.0f;
    #pragma unroll
    for (int v = 0; v < 8; ++v) {
      float l = __fmaf_rn(2.0f, acc[v], -cn) * inv_scale;
      psum = __fmaf_rn(__expf(l), Zl[v], psum);
    }
    // Both halves target the same kcol (different row sets): 2-way ds merge.
    atomicAdd(&lds_soft[kcol], psum);
  }

  __syncthreads();
  for (int i = tid; i < KCB; i += 256) atomicAdd(&soft_g[i], lds_soft[i]);
}

// ---------------------------------------------------------------------------
// Kernel 3: gather quantized vectors (exact f32), codes, segment sums.
// ---------------------------------------------------------------------------
__global__ __launch_bounds__(256)
void k_gather(const float* __restrict__ X, const float* __restrict__ CB,
              const int* __restrict__ codes, float* __restrict__ q_out,
              float* __restrict__ codes_out, float* __restrict__ bcount,
              float* __restrict__ bweight) {
  int n = blockIdx.x * 2 + (threadIdx.x >> 7);
  int d = threadIdx.x & 127;
  int c = codes[n];
  q_out[(size_t)n * DIM + d] = CB[(size_t)c * DIM + d];
  atomicAdd(&bweight[(size_t)c * DIM + d], X[(size_t)n * DIM + d]);
  if (d == 0) {
    codes_out[n] = (float)c;
    atomicAdd(&bcount[c], 1.0f);
  }
}

// ---------------------------------------------------------------------------
// Kernel 4: new_count, soft_usage output, grand total (for smoothing).
// ---------------------------------------------------------------------------
__global__ __launch_bounds__(256)
void k_count(const float* __restrict__ ema_count, const float* __restrict__ bcount,
             const float* __restrict__ soft_ws, float* __restrict__ out_count,
             float* __restrict__ out_soft, float* __restrict__ total) {
  int k = blockIdx.x * 256 + threadIdx.x;
  float nc = DECAYF * ema_count[k] + ONEMD * bcount[k];
  out_count[k] = nc;
  out_soft[k]  = soft_ws[k] * (1.0f / (float)N_TOK);
  float s = nc;
  #pragma unroll
  for (int m = 16; m >= 1; m >>= 1) s += __shfl_xor(s, m, 32);
  __shared__ float red[8];
  if ((threadIdx.x & 31) == 0) red[threadIdx.x >> 5] = s;
  __syncthreads();
  if (threadIdx.x < 8) {
    float r = red[threadIdx.x];
    #pragma unroll
    for (int m = 4; m >= 1; m >>= 1) r += __shfl_xor(r, m, 8);
    if (threadIdx.x == 0) atomicAdd(total, r);
  }
}

// ---------------------------------------------------------------------------
// Kernel 5: new_weight, smoothed counts, new_codebook.
// ---------------------------------------------------------------------------
__global__ __launch_bounds__(256)
void k_final(const float* __restrict__ ema_weight, const float* __restrict__ bweight,
             const float* __restrict__ out_count, const float* __restrict__ total,
             float* __restrict__ out_weight, float* __restrict__ out_cb) {
  size_t i = (size_t)blockIdx.x * 256 + threadIdx.x;
  int k = (int)(i >> 7);
  float nw = DECAYF * ema_weight[i] + ONEMD * bweight[i];
  out_weight[i] = nw;
  float T  = *total;
  float nc = out_count[k];
  float sm = (nc + EPSF) / (T + (float)KCB * EPSF) * T;
  out_cb[i] = nw / fmaxf(sm, EPSF);
}

// ---------------------------------------------------------------------------
extern "C" void kernel_launch(void* const* d_in, const int* in_sizes, int n_in,
                              void* d_out, int out_size, void* d_ws, size_t ws_size,
                              hipStream_t stream) {
  (void)in_sizes; (void)n_in; (void)out_size; (void)ws_size;
  const float* X  = (const float*)d_in[0];  // [N, D]
  const float* CB = (const float*)d_in[1];  // [K, D]
  const float* EW = (const float*)d_in[2];  // [K, D]
  const float* EC = (const float*)d_in[3];  // [K]

  float* out = (float*)d_out;
  float* q_out     = out;                         // [N*D] quantized
  float* codes_out = out + (size_t)N_TOK * DIM;   // [N]   codes (as float)
  float* soft_out  = codes_out + N_TOK;           // [K]   soft_usage
  float* cb_out    = soft_out + KCB;              // [K*D] new_codebook
  float* cnt_out   = cb_out + (size_t)KCB * DIM;  // [K]   new_count
  float* w_out     = cnt_out + KCB;               // [K*D] new_weight

  float* ws        = (float*)d_ws;
  int*   ws_codes  = (int*)ws;                    // [N] int codes
  float* ws_cnorm  = ws + 32768;                  // [K]
  float* ws_soft   = ws + 36864;                  // [K] accumulator
  float* ws_bcount = ws + 40960;                  // [K] accumulator
  float* ws_total  = ws + 45056;                  // [1]
  float* ws_bwt    = ws + 45312;                  // [K*D] accumulator
  _Float16* ws_xh  = (_Float16*)(ws + 569600);    // [N*D] f16 X (16B aligned)
  _Float16* ws_ch  = ws_xh + (size_t)N_TOK * DIM; // [K*D] f16 codebook

  k_cvt   <<<(N_TOK * DIM / 4) / 256, 256, 0, stream>>>(X, ws_xh);
  k_cvt   <<<(KCB  * DIM / 4) / 256, 256, 0, stream>>>(CB, ws_ch);
  k_init  <<<KCB,             128, 0, stream>>>(CB, ws_cnorm, ws_soft, ws_bcount, ws_bwt, ws_total);
  k_main  <<<N_TOK / ROWS_WG, 256, 0, stream>>>(ws_xh, ws_ch, ws_cnorm, ws_codes, ws_soft);
  k_gather<<<N_TOK / 2,       256, 0, stream>>>(X, CB, ws_codes, q_out, codes_out, ws_bcount, ws_bwt);
  k_count <<<KCB / 256,       256, 0, stream>>>(EC, ws_bcount, ws_soft, cnt_out, soft_out, ws_total);
  k_final <<<(KCB * DIM) / 256, 256, 0, stream>>>(EW, ws_bwt, cnt_out, ws_total, w_out, cb_out);
}
